// TFDebertaAttention_33054068310414
// MI455X (gfx1250) — matrix operand order repaired and checked by reading, not compiled, not run on hardware
//
#include <hip/hip_runtime.h>
#include <hip/hip_bf16.h>
#include <math.h>

// ---------------------------------------------------------------------------
// DeBERTa disentangled attention block for MI455X (gfx1250), wave32 + WMMA.
// B=4, S=1024, H=1024, NH=16, D=64, span=512, scale=sqrt(192).
// Pipeline:
//   K1 qkv_kernel : hidden @ in_proj_w -> qh(bf16, scaled+bias), kT(bf16, [b,h,D,S]), vh(bf16,+bias)
//   K2 pos_kernel : rel @ pos_proj_w -> pkT(bf16,[h,D,P]); rel @ pos_q_w + b -> pq(bf16,[h,P,D], scaled)
//   K3 attn_kernel: fused scores(qk + banded c2p/p2c gather) + online softmax + PV -> ctx f32
//                   k^T / V tiles staged in LDS via GLOBAL_LOAD_ASYNC_TO_LDS_B128
//   K4 proj_kernel: ctx @ out_w + out_b + hidden -> ybuf f32
//   K5 ln_kernel  : LayerNorm(ybuf) -> out f32
// ---------------------------------------------------------------------------

typedef __bf16 bf16_t;
typedef __attribute__((ext_vector_type(16))) __bf16 v16bf;
typedef __attribute__((ext_vector_type(8)))  float  v8f;

#define WMMA_BF16(a, b, c) \
  __builtin_amdgcn_wmma_f32_16x16x32_bf16(false, (a), false, (b), (short)0, (c), false, false)

constexpr int   Bb = 4, Ss = 1024, Hh = 1024, NHn = 16, Dd = 64, SPAN = 512, P2 = 1024;
constexpr float INV_SCALE = 0.07216878364870323f;  // 1/sqrt(64*3)

__device__ inline v8f zero8() {
  v8f z = {0.f, 0.f, 0.f, 0.f, 0.f, 0.f, 0.f, 0.f};
  return z;
}

// Async copy of 16 bytes global -> LDS (per lane), tracked by ASYNCcnt.
__device__ inline void async_copy_b128(const void* gptr, void* lptr) {
  unsigned lds_off = (unsigned)(unsigned long long)lptr;  // low 32 bits of generic LDS addr = LDS byte offset
  unsigned long long gaddr = (unsigned long long)gptr;
  asm volatile("global_load_async_to_lds_b128 %0, %1, off"
               :
               : "v"(lds_off), "v"(gaddr)
               : "memory");
}
__device__ inline void wait_asynccnt0() {
  asm volatile("s_wait_asynccnt 0x0" ::: "memory");
}

// A-fragment (16x32, bf16): lane = M row; halves 0-7 -> K kb..kb+7, 8-15 -> K kb+16..kb+23
__device__ inline v16bf load_A_f32(const float* __restrict__ base, int ld, int lane) {
  int m = lane & 15;
  int kb = (lane >> 4) << 3;
  const float* r = base + m * ld + kb;
  v16bf a;
#pragma unroll
  for (int i = 0; i < 8; i++) { a[i] = (__bf16)r[i]; a[i + 8] = (__bf16)r[i + 16]; }
  return a;
}
__device__ inline v16bf load_A_bf16(const bf16_t* __restrict__ base, int ld, int lane) {
  int m = lane & 15;
  int kb = (lane >> 4) << 3;
  const bf16_t* r = base + m * ld + kb;
  v16bf a;
#pragma unroll
  for (int i = 0; i < 8; i++) { a[i] = r[i]; a[i + 8] = r[i + 16]; }
  return a;
}
// B-fragment (32x16, bf16): lane = N col; halves 0-7 -> K kb..kb+7, 8-15 -> K kb+16..kb+23
__device__ inline v16bf load_B_f32(const float* __restrict__ base, int ld, int lane) {
  int kb = (lane >> 4) << 3;
  const float* r = base + (lane & 15);
  v16bf b;
#pragma unroll
  for (int i = 0; i < 8; i++) { b[i] = (__bf16)r[(kb + i) * ld]; b[i + 8] = (__bf16)r[(kb + 16 + i) * ld]; }
  return b;
}
__device__ inline v16bf load_B_bf16(const bf16_t* __restrict__ base, int ld, int lane) {
  int kb = (lane >> 4) << 3;
  const bf16_t* r = base + (lane & 15);
  v16bf b;
#pragma unroll
  for (int i = 0; i < 8; i++) { b[i] = r[(kb + i) * ld]; b[i + 8] = r[(kb + 16 + i) * ld]; }
  return b;
}

__device__ inline float redmax16(float v) {
#pragma unroll
  for (int m = 1; m < 16; m <<= 1) v = fmaxf(v, __shfl_xor(v, m, 32));
  return v;
}
__device__ inline float redsum16(float v) {
#pragma unroll
  for (int m = 1; m < 16; m <<= 1) v += __shfl_xor(v, m, 32);
  return v;
}

// ---------------- K1: QKV projection (4096 x 3072 x 1024) ----------------
__global__ __launch_bounds__(256) void qkv_kernel(
    const float* __restrict__ hid, const float* __restrict__ w,
    const float* __restrict__ qb, const float* __restrict__ vb,
    bf16_t* __restrict__ qh, bf16_t* __restrict__ kT, bf16_t* __restrict__ vh) {
  int lane = threadIdx.x & 31, wv = threadIdx.x >> 5;
  int m0 = blockIdx.x * 128 + wv * 16;
  int n0 = blockIdx.y * 64;
  v8f c[4];
#pragma unroll
  for (int t = 0; t < 4; t++) c[t] = zero8();
  for (int k0 = 0; k0 < Hh; k0 += 32) {
    v16bf a = load_A_f32(hid + m0 * Hh + k0, Hh, lane);
#pragma unroll
    for (int t = 0; t < 4; t++) {
      v16bf b = load_B_f32(w + k0 * 3072 + n0 + 16 * t, 3072, lane);
      c[t] = WMMA_BF16(a, b, c[t]);
    }
  }
  int hs = (lane >> 4) << 3;
#pragma unroll
  for (int t = 0; t < 4; t++) {
    int ncol = n0 + 16 * t + (lane & 15);
    int h = ncol / 192, r = ncol - h * 192, sel = r >> 6, d = r & 63;
    int hd = (h << 6) + d;
#pragma unroll
    for (int j = 0; j < 8; j++) {
      int m = m0 + j + hs;
      int bidx = m >> 10, s = m & 1023;
      int bh = bidx * NHn + h;
      float cv = c[t][j];
      if (sel == 0)      qh[(bh * Ss + s) * Dd + d] = (__bf16)((cv + qb[hd]) * INV_SCALE);
      else if (sel == 1) kT[(bh * Dd + d) * Ss + s] = (__bf16)cv;
      else               vh[(bh * Ss + s) * Dd + d] = (__bf16)(cv + vb[hd]);
    }
  }
}

// ---------------- K2: positional projections (1024 x 1024 x 1024) ----------------
__global__ __launch_bounds__(256) void pos_kernel(
    const float* __restrict__ re, const float* __restrict__ w,
    const float* __restrict__ bias, bf16_t* __restrict__ out, int mode, float scl) {
  int lane = threadIdx.x & 31, wv = threadIdx.x >> 5;
  int m0 = blockIdx.x * 128 + wv * 16;
  int n0 = blockIdx.y * 64;
  v8f c[4];
#pragma unroll
  for (int t = 0; t < 4; t++) c[t] = zero8();
  for (int k0 = 0; k0 < Hh; k0 += 32) {
    v16bf a = load_A_f32(re + m0 * Hh + k0, Hh, lane);
#pragma unroll
    for (int t = 0; t < 4; t++) {
      v16bf b = load_B_f32(w + k0 * Hh + n0 + 16 * t, Hh, lane);
      c[t] = WMMA_BF16(a, b, c[t]);
    }
  }
  int hs = (lane >> 4) << 3;
#pragma unroll
  for (int t = 0; t < 4; t++) {
    int ncol = n0 + 16 * t + (lane & 15);
#pragma unroll
    for (int j = 0; j < 8; j++) {
      int p = m0 + j + hs;
      float cv = c[t][j];
      if (mode == 0) {
        out[ncol * P2 + p] = (__bf16)cv;                 // pos_key^T : [h*64+d][p]
      } else {
        int h = ncol >> 6, d = ncol & 63;
        out[(h * P2 + p) * Dd + d] = (__bf16)((cv + bias[ncol]) * scl);  // pos_q : [h][p][d]
      }
    }
  }
}

// ---------------- K3: fused disentangled attention ----------------
__global__ __launch_bounds__(256) void attn_kernel(
    const bf16_t* __restrict__ qh, const bf16_t* __restrict__ kT,
    const bf16_t* __restrict__ vh, const bf16_t* __restrict__ pkT,
    const bf16_t* __restrict__ pq, const unsigned char* __restrict__ mask,
    float* __restrict__ ctx) {
  __shared__ bf16_t ls_c2p[8][16 * 48];   // 12 KB  per-wave c2p gather scratch
  __shared__ bf16_t ls_p2c[8][48 * 16];   // 12 KB  per-wave p2c gather scratch
  __shared__ bf16_t ls_pr[8][16 * 32];    //  8 KB  per-wave probs transpose
  __shared__ bf16_t ktile[64 * 32];       //  4 KB  k^T tile [d=0..63][k=kc..kc+31]
  __shared__ bf16_t vtile[32 * 64];       //  4 KB  v  tile [k=kc..kc+31][d=0..63]

  int lane = threadIdx.x & 31, wv = threadIdx.x >> 5;
  int tid = threadIdx.x;
  int bh = blockIdx.x >> 3;
  int qt = blockIdx.x & 7;
  int b = bh >> 4, h = bh & 15;
  int q0 = qt * 128 + wv * 16;
  int hs = (lane >> 4) << 3;
  int n = lane & 15;

  const bf16_t* qbase = qh + (bh * Ss + q0) * Dd;
  v16bf aq0 = load_A_bf16(qbase, Dd, lane);
  v16bf aq1 = load_A_bf16(qbase + 32, Dd, lane);

  v8f acc[4];
#pragma unroll
  for (int t = 0; t < 4; t++) acc[t] = zero8();
  float rowmax[8], rowsum[8];
#pragma unroll
  for (int j = 0; j < 8; j++) { rowmax[j] = -1e30f; rowsum[j] = 0.f; }

  // per-thread staging segments (16B each): k^T tile 64x32, v tile 32x64
  int kt_row = tid >> 2, kt_seg = (tid & 3) * 8;
  int vt_row = tid >> 3, vt_seg = (tid & 7) * 8;
  const bf16_t* kT_bh = kT + bh * Dd * Ss;
  const bf16_t* vh_bh = vh + bh * Ss * Dd;

  for (int kc = 0; kc < Ss; kc += 32) {
    // ---- stage k^T / v tiles into LDS with async DMA ----
    __syncthreads();  // previous chunk's tile readers are done
    async_copy_b128(kT_bh + kt_row * Ss + kc + kt_seg, &ktile[kt_row * 32 + kt_seg]);
    async_copy_b128(vh_bh + (kc + vt_row) * Dd + vt_seg, &vtile[vt_row * 64 + vt_seg]);
    if (kc + 32 < Ss) __builtin_prefetch(kT_bh + kt_row * Ss + kc + 32 + kt_seg, 0, 1);
    wait_asynccnt0();
    __syncthreads();  // tiles visible to all waves

    v8f sfr[2];
#pragma unroll 1
    for (int jt = 0; jt < 2; jt++) {
      int k0 = kc + 16 * jt;
      v16bf bk0 = load_B_bf16(&ktile[jt * 16], 32, lane);            // d = 0..31
      v16bf bk1 = load_B_bf16(&ktile[32 * 32 + jt * 16], 32, lane);  // d = 32..63

      v8f s = zero8();
      s = WMMA_BF16(aq0, bk0, s);
      s = WMMA_BF16(aq1, bk1, s);

      // relative-position band: p = clip(q - k + span) over this 16x16 tile
      int cd = q0 - k0 + SPAN;
      int pb = cd - 15;
      pb = pb < 0 ? 0 : (pb > (P2 - 48) ? (P2 - 48) : pb);
      int pb16 = pb & ~15;

      v8f cc[3], pc[3];
#pragma unroll
      for (int t = 0; t < 3; t++) { cc[t] = zero8(); pc[t] = zero8(); }
#pragma unroll
      for (int t = 0; t < 3; t++) {  // c2p: q (16xD) x pos_key^T (Dx48)
        v16bf b0 = load_B_bf16(pkT + (h * Dd + 0) * P2 + pb16 + 16 * t, P2, lane);
        v16bf b1 = load_B_bf16(pkT + (h * Dd + 32) * P2 + pb16 + 16 * t, P2, lane);
        cc[t] = WMMA_BF16(aq0, b0, cc[t]);
        cc[t] = WMMA_BF16(aq1, b1, cc[t]);
      }
#pragma unroll
      for (int t = 0; t < 3; t++) {  // p2c: pos_q (48xD) x k^T (Dx16)
        const bf16_t* pqb_ = pq + (h * P2 + pb16 + 16 * t) * Dd;
        v16bf a0 = load_A_bf16(pqb_, Dd, lane);
        v16bf a1 = load_A_bf16(pqb_ + 32, Dd, lane);
        pc[t] = WMMA_BF16(a0, bk0, pc[t]);
        pc[t] = WMMA_BF16(a1, bk1, pc[t]);
      }

      __syncthreads();  // protect gather scratch from previous readers
#pragma unroll
      for (int t = 0; t < 3; t++)
#pragma unroll
        for (int j = 0; j < 8; j++) {
          int m = j + hs;
          ls_c2p[wv][m * 48 + t * 16 + n] = (__bf16)cc[t][j];
          ls_p2c[wv][(t * 16 + m) * 16 + n] = (__bf16)pc[t][j];
        }
      __syncthreads();

#pragma unroll
      for (int j = 0; j < 8; j++) {
        int m = j + hs;
        int p = q0 + m - (k0 + n) + SPAN;
        p = p < 0 ? 0 : (p > P2 - 1 ? P2 - 1 : p);
        int idx = p - pb16;
        float add = (float)ls_c2p[wv][m * 48 + idx] + (float)ls_p2c[wv][idx * 16 + n];
        unsigned char mk = mask[(b * Ss + q0 + m) * Ss + k0 + n];
        float sv = s[j] + add;
        s[j] = mk ? sv : -1e30f;
      }
      sfr[jt] = s;
    }

    // ---- online softmax over this 32-col chunk ----
    float pr0[8], pr1[8], corr[8];
#pragma unroll
    for (int j = 0; j < 8; j++) {
      float mx = redmax16(fmaxf(sfr[0][j], sfr[1][j]));
      float mnew = fmaxf(rowmax[j], mx);
      float co = __expf(rowmax[j] - mnew);
      float p0 = __expf(sfr[0][j] - mnew);
      float p1 = __expf(sfr[1][j] - mnew);
      rowsum[j] = rowsum[j] * co + redsum16(p0 + p1);
      rowmax[j] = mnew;
      corr[j] = co;
      pr0[j] = p0; pr1[j] = p1;
    }
#pragma unroll
    for (int t = 0; t < 4; t++)
#pragma unroll
      for (int j = 0; j < 8; j++) acc[t][j] *= corr[j];

    __syncthreads();  // transpose probs C-layout -> row-major bf16 via LDS
#pragma unroll
    for (int j = 0; j < 8; j++) {
      int m = j + hs;
      ls_pr[wv][m * 32 + n] = (__bf16)pr0[j];
      ls_pr[wv][m * 32 + 16 + n] = (__bf16)pr1[j];
    }
    __syncthreads();
    v16bf ap = load_A_bf16(&ls_pr[wv][0], 32, lane);
#pragma unroll
    for (int t = 0; t < 4; t++) {
      v16bf bv = load_B_bf16(&vtile[16 * t], 64, lane);
      acc[t] = WMMA_BF16(ap, bv, acc[t]);
    }
  }

#pragma unroll
  for (int j = 0; j < 8; j++) {
    int m = j + hs;
    float inv = 1.0f / rowsum[j];
#pragma unroll
    for (int t = 0; t < 4; t++) {
      int d = 16 * t + n;
      ctx[(b * Ss + q0 + m) * Hh + (h << 6) + d] = acc[t][j] * inv;
    }
  }
}

// ---------------- K4: output projection + residual ----------------
__global__ __launch_bounds__(256) void proj_kernel(
    const float* __restrict__ ctxb, const float* __restrict__ w,
    const float* __restrict__ ob, const float* __restrict__ hid,
    float* __restrict__ y) {
  int lane = threadIdx.x & 31, wv = threadIdx.x >> 5;
  int m0 = blockIdx.x * 128 + wv * 16;
  int n0 = blockIdx.y * 64;
  v8f c[4];
#pragma unroll
  for (int t = 0; t < 4; t++) c[t] = zero8();
  for (int k0 = 0; k0 < Hh; k0 += 32) {
    v16bf a = load_A_f32(ctxb + m0 * Hh + k0, Hh, lane);
#pragma unroll
    for (int t = 0; t < 4; t++) {
      v16bf b = load_B_f32(w + k0 * Hh + n0 + 16 * t, Hh, lane);
      c[t] = WMMA_BF16(a, b, c[t]);
    }
  }
  int hs = (lane >> 4) << 3;
#pragma unroll
  for (int t = 0; t < 4; t++) {
    int ncol = n0 + 16 * t + (lane & 15);
#pragma unroll
    for (int j = 0; j < 8; j++) {
      int m = m0 + j + hs;
      y[m * Hh + ncol] = c[t][j] + ob[ncol] + hid[m * Hh + ncol];
    }
  }
}

// ---------------- K5: LayerNorm ----------------
__global__ __launch_bounds__(256) void ln_kernel(
    const float* __restrict__ y, const float* __restrict__ g,
    const float* __restrict__ be, float* __restrict__ out) {
  __shared__ float red[18];
  int row = blockIdx.x;
  const float* yr = y + row * Hh;
  float v[4];
  float s = 0.f, s2 = 0.f;
#pragma unroll
  for (int i = 0; i < 4; i++) {
    float x = yr[threadIdx.x + 256 * i];
    v[i] = x; s += x; s2 += x * x;
  }
#pragma unroll
  for (int m = 1; m < 32; m <<= 1) { s += __shfl_xor(s, m, 32); s2 += __shfl_xor(s2, m, 32); }
  int wv = threadIdx.x >> 5, lane = threadIdx.x & 31;
  if (lane == 0) { red[wv] = s; red[8 + wv] = s2; }
  __syncthreads();
  if (threadIdx.x == 0) {
    float ts = 0.f, t2 = 0.f;
    for (int i = 0; i < 8; i++) { ts += red[i]; t2 += red[8 + i]; }
    red[16] = ts; red[17] = t2;
  }
  __syncthreads();
  float mean = red[16] * (1.0f / Hh);
  float var = red[17] * (1.0f / Hh) - mean * mean;
  float rstd = rsqrtf(var + 1e-7f);
#pragma unroll
  for (int i = 0; i < 4; i++) {
    int col = threadIdx.x + 256 * i;
    out[row * Hh + col] = g[col] * (v[i] - mean) * rstd + be[col];
  }
}

// ---------------- launcher ----------------
extern "C" void kernel_launch(void* const* d_in, const int* in_sizes, int n_in,
                              void* d_out, int out_size, void* d_ws, size_t ws_size,
                              hipStream_t stream) {
  const float* hid = (const float*)d_in[0];
  const unsigned char* mask = (const unsigned char*)d_in[1];
  const float* rel = (const float*)d_in[2];
  const float* in_w = (const float*)d_in[3];
  const float* qb = (const float*)d_in[4];
  const float* vb = (const float*)d_in[5];
  const float* pkw = (const float*)d_in[6];
  const float* pqw = (const float*)d_in[7];
  const float* pqb = (const float*)d_in[8];
  const float* ow = (const float*)d_in[9];
  const float* ob = (const float*)d_in[10];
  const float* lng = (const float*)d_in[11];
  const float* lnb = (const float*)d_in[12];
  float* out = (float*)d_out;

  char* ws = (char*)d_ws;
  bf16_t* qh  = (bf16_t*)(ws);                          // 8 MB  [b,h,S,D]
  bf16_t* kT  = (bf16_t*)(ws + (size_t)(8u << 20));     // 8 MB  [b,h,D,S]
  bf16_t* vh  = (bf16_t*)(ws + (size_t)(16u << 20));    // 8 MB  [b,h,S,D]
  bf16_t* pkT = (bf16_t*)(ws + (size_t)(24u << 20));    // 2 MB  [h,D,P]
  bf16_t* pq  = (bf16_t*)(ws + (size_t)(26u << 20));    // 2 MB  [h,P,D]
  float*  ctx = (float*)(ws + (size_t)(28u << 20));     // 16 MB [b,S,H]
  float*  ybf = (float*)(ws + (size_t)(44u << 20));     // 16 MB [b,S,H]

  qkv_kernel<<<dim3(32, 48), 256, 0, stream>>>(hid, in_w, qb, vb, qh, kT, vh);
  pos_kernel<<<dim3(8, 16), 256, 0, stream>>>(rel, pkw, pqb, pkT, 0, 1.0f);
  pos_kernel<<<dim3(8, 16), 256, 0, stream>>>(rel, pqw, pqb, pq, 1, INV_SCALE);
  attn_kernel<<<dim3(512), 256, 0, stream>>>(qh, kT, vh, pkT, pq, mask, ctx);
  proj_kernel<<<dim3(32, 16), 256, 0, stream>>>(ctx, ow, ob, hid, ybf);
  ln_kernel<<<dim3(4096), 256, 0, stream>>>(ybf, lng, lnb, out);
}
